// NeuralNet_44092134261082
// MI455X (gfx1250) — matrix operand order repaired and checked by reading, not compiled, not run on hardware
//
#include <hip/hip_runtime.h>

typedef __attribute__((ext_vector_type(16))) _Float16 v16h;
typedef __attribute__((ext_vector_type(8)))  _Float16 v8h;
typedef __attribute__((ext_vector_type(8)))  float    v8f;
typedef __attribute__((ext_vector_type(4)))  float    v4f;

#define NN   1024
#define CC   8
#define DIMD 128
#define MROW 8192   // N*C
#define RSQRT_D 0.08838834764831845f  // 1/sqrt(128)

// ---------- WMMA helpers ----------
__device__ __forceinline__ v8f wmma_f16(v16h a, v16h b, v8f c) {
    // D = A(16x32 f16) x B(32x16 f16) + C(16x16 f32)
    return __builtin_amdgcn_wmma_f32_16x16x32_f16(
        /*neg_a=*/false, a, /*neg_b=*/false, b,
        /*c_mod=*/(short)0, c, /*reuse_a=*/false, /*reuse_b=*/false);
}

// A-matrix 16x32 f16 per-lane fragment: row = base+(lane&15),
// halves at K = ks+8h and K = ks+16+8h  (ISA 7.12.2 16-bit A layout)
__device__ __forceinline__ v16h load_a(const _Float16* __restrict__ row, int ks, int h) {
    union { v16h v; v8h p[2]; } u;
    u.p[0] = *(const v8h*)(row + ks + 8 * h);
    u.p[1] = *(const v8h*)(row + ks + 16 + 8 * h);
    return u.v;
}

// B-matrix 32x16 f16 per-lane fragment: col = base+(lane&15),
// halves at K = ks+8h and K = ks+16+8h (same order as A)
__device__ __forceinline__ v16h load_b(const _Float16* __restrict__ row, int ks, int h) {
    union { v16h v; v8h p[2]; } u;
    u.p[0] = *(const v8h*)(row + ks + 8 * h);
    u.p[1] = *(const v8h*)(row + ks + 16 + 8 * h);
    return u.v;
}

// ---------- prep kernels ----------
__global__ __launch_bounds__(256) void k_zero(float* __restrict__ p) {
    size_t tid = (size_t)blockIdx.x * blockDim.x + threadIdx.x;  // 2M threads
    v4f z = {0.f, 0.f, 0.f, 0.f};
    *(v4f*)(p + tid * 4) = z;
}

__global__ __launch_bounds__(256) void k_prepE(const float* __restrict__ E,
                                               const float* __restrict__ R,
                                               _Float16* __restrict__ Eh,
                                               _Float16* __restrict__ Ah) {
    size_t idx = (size_t)blockIdx.x * blockDim.x + threadIdx.x;  // < 8192*128
    int d = (int)(idx & 127);
    float e = E[idx];
    Eh[idx] = (_Float16)e;
#pragma unroll
    for (int k = 0; k < 3; ++k)
        Ah[(size_t)k * (MROW * DIMD) + idx] = (_Float16)(e * R[k * DIMD + d]);
}

__global__ __launch_bounds__(256) void k_prepF(const float* __restrict__ F,
                                               const float* __restrict__ D,
                                               _Float16* __restrict__ Fh,
                                               _Float16* __restrict__ FDh) {
    size_t idx = (size_t)blockIdx.x * blockDim.x + threadIdx.x;  // < 1024*128
    int d = (int)(idx & 127);
    float f = F[idx];
    Fh[idx] = (_Float16)f;
#pragma unroll
    for (int k = 0; k < 3; ++k)
        FDh[(size_t)k * (NN * DIMD) + idx] = (_Float16)(f * D[k * DIMD + d] * RSQRT_D);
}

__global__ __launch_bounds__(256) void k_psi(const float* __restrict__ E,
                                             const float* __restrict__ B,
                                             const float* __restrict__ Fc,
                                             float* __restrict__ psi) {
    int tid = blockIdx.x * blockDim.x + threadIdx.x;  // 0..8191  (i*8+c)
    int i = tid >> 3;
    const float* e  = E  + (size_t)tid * DIMD;
    const float* fc = Fc + (size_t)i * DIMD;
    float s = 0.f;
    for (int d = 0; d < DIMD; ++d) s += e[d] * B[d] * fc[d];
    psi[tid] = s;
}

// ---------- logits + softmax_k -> a[i][j][4]  (WMMA, M=N=1024, K=128) ----------
__global__ __launch_bounds__(256) void k_logits(const _Float16* __restrict__ FDh,
                                                const _Float16* __restrict__ Fh,
                                                float* __restrict__ aW) {
    int wave = (blockIdx.x * blockDim.x + threadIdx.x) >> 5;  // 0..1023
    int lane = threadIdx.x & 31;
    int wm = wave >> 5, wn = wave & 31;     // 32x32 wave grid, 32x32 out per wave
    int lr = lane & 15, h = lane >> 4;

    const _Float16* ar[3][2];
    const _Float16* br[2];
#pragma unroll
    for (int mt = 0; mt < 2; ++mt) {
        int row = wm * 32 + mt * 16 + lr;
#pragma unroll
        for (int k = 0; k < 3; ++k) ar[k][mt] = FDh + ((size_t)k * NN + row) * DIMD;
    }
#pragma unroll
    for (int nt = 0; nt < 2; ++nt) br[nt] = Fh + (size_t)(wn * 32 + nt * 16 + lr) * DIMD;

    v8f acc[3][2][2] = {};
#pragma unroll
    for (int ks = 0; ks < DIMD; ks += 32) {
        v16h b0 = load_b(br[0], ks, h);
        v16h b1 = load_b(br[1], ks, h);
#pragma unroll
        for (int k = 0; k < 3; ++k) {
#pragma unroll
            for (int mt = 0; mt < 2; ++mt) {
                v16h a = load_a(ar[k][mt], ks, h);
                acc[k][mt][0] = wmma_f16(a, b0, acc[k][mt][0]);
                acc[k][mt][1] = wmma_f16(a, b1, acc[k][mt][1]);
            }
        }
    }
#pragma unroll
    for (int mt = 0; mt < 2; ++mt) {
#pragma unroll
        for (int nt = 0; nt < 2; ++nt) {
            int j = wn * 32 + nt * 16 + lr;
#pragma unroll
            for (int r = 0; r < 8; ++r) {
                int i = wm * 32 + mt * 16 + 8 * h + r;
                float l0 = acc[0][mt][nt][r], l1 = acc[1][mt][nt][r], l2 = acc[2][mt][nt][r];
                float mx = fmaxf(l0, fmaxf(l1, l2));
                float e0 = __expf(l0 - mx), e1 = __expf(l1 - mx), e2 = __expf(l2 - mx);
                float inv = 1.0f / (e0 + e1 + e2);
                v4f o; o[0] = e0 * inv; o[1] = e1 * inv; o[2] = e2 * inv; o[3] = 0.f;
                *(v4f*)(aW + ((size_t)i * NN + j) * 4) = o;
            }
        }
    }
}

// ---------- phi GEMM: M=N=8192, K=128, 3 accumulator sets, weighted by a ----------
// phi stored as [i][j][cj][ci] f16 so each lane's 8 D-elements (ci=0..7) are one v8h store.
__global__ __launch_bounds__(256) void k_phi(const _Float16* __restrict__ Ah,
                                             const _Float16* __restrict__ Eh,
                                             const float* __restrict__ aW,
                                             _Float16* __restrict__ phi) {
    int wave = (blockIdx.x * blockDim.x + threadIdx.x) >> 5;  // 0..65535
    int lane = threadIdx.x & 31;
    int wm = wave >> 8, wn = wave & 255;    // 256x256 wave grid, 32x32 out per wave
    int lr = lane & 15, h = lane >> 4;

    const _Float16* ar[3][2];
    const _Float16* br[2];
#pragma unroll
    for (int mt = 0; mt < 2; ++mt) {
        int row = wm * 32 + mt * 16 + lr;
#pragma unroll
        for (int k = 0; k < 3; ++k) ar[k][mt] = Ah + ((size_t)k * MROW + row) * DIMD;
    }
#pragma unroll
    for (int nt = 0; nt < 2; ++nt) br[nt] = Eh + (size_t)(wn * 32 + nt * 16 + lr) * DIMD;

    v8f acc[3][2][2] = {};
#pragma unroll
    for (int ks = 0; ks < DIMD; ks += 32) {
        v16h b0 = load_b(br[0], ks, h);
        v16h b1 = load_b(br[1], ks, h);
#pragma unroll
        for (int k = 0; k < 3; ++k) {
#pragma unroll
            for (int mt = 0; mt < 2; ++mt) {
                v16h a = load_a(ar[k][mt], ks, h);
                acc[k][mt][0] = wmma_f16(a, b0, acc[k][mt][0]);
                acc[k][mt][1] = wmma_f16(a, b1, acc[k][mt][1]);
            }
        }
    }
    // epilogue: per lane, per (mt,nt): fixed (i, j, cj), elements r = ci
#pragma unroll
    for (int mt = 0; mt < 2; ++mt) {
        int i = wm * 4 + mt * 2 + h;          // (wm*32 + mt*16 + 8h) >> 3
#pragma unroll
        for (int nt = 0; nt < 2; ++nt) {
            int j  = wn * 4 + nt * 2 + (lr >> 3);
            int cj = lr & 7;
            v4f av = *(const v4f*)(aW + ((size_t)i * NN + j) * 4);
            v8h o;
#pragma unroll
            for (int r = 0; r < 8; ++r) {
                float v = av[0] * acc[0][mt][nt][r]
                        + av[1] * acc[1][mt][nt][r]
                        + av[2] * acc[2][mt][nt][r];
                o[r] = (_Float16)v;
            }
            *(v8h*)(phi + (((size_t)i * NN + j) * 8 + cj) * 8) = o;
        }
    }
}

// ---------- S[i][c] = psi[i][c] + sum_src mbar[src][i][c] ----------
__global__ __launch_bounds__(256) void k_reduceS(const float* __restrict__ mbar,
                                                 const float* __restrict__ psi,
                                                 float* __restrict__ S) {
    __shared__ float red[256];
    int t  = threadIdx.x;
    int ii = t >> 6;          // 0..3 (dst row within block)
    int g  = (t >> 3) & 7;    // 0..7 (src group)
    int c  = t & 7;
    int i  = blockIdx.x * 4 + ii;
    float p = 0.f;
    for (int s = 0; s < 128; ++s) {
        int src = g + s * 8;
        p += mbar[((size_t)src * NN + i) * 8 + c];
    }
    red[t] = p;
    __syncthreads();
    if (g == 0) {
        float tot = p;
#pragma unroll
        for (int gg = 1; gg < 8; ++gg) tot += red[(ii << 6) + (gg << 3) + c];
        S[i * 8 + c] = psi[i * 8 + c] + tot;
    }
}

// ---------- LBP update: one thread per (i,j) ----------
__global__ __launch_bounds__(256) void k_update(const float* __restrict__ mcur,
                                                const float* __restrict__ S,
                                                const _Float16* __restrict__ phi,
                                                float* __restrict__ mnext) {
    int tid = blockIdx.x * blockDim.x + threadIdx.x;  // 0..1048575
    int i = tid >> 10;
    int j = tid & 1023;
    float s[8];
    const float* Si = S + i * 8;
    const float* mt = mcur + ((size_t)j * NN + i) * 8;   // mbar[j,i,:]
#pragma unroll
    for (int c = 0; c < 8; ++c) s[c] = Si[c] - mt[c];

    const _Float16* pb = phi + ((size_t)i * NN + j) * 64;  // layout [cj][ci]
    float msg[8];
#pragma unroll
    for (int cj = 0; cj < 8; ++cj) {
        v8h ph = *(const v8h*)(pb + cj * 8);
        float m = s[0] + (float)ph[0];
#pragma unroll
        for (int ci = 1; ci < 8; ++ci) m = fmaxf(m, s[ci] + (float)ph[ci]);
        msg[cj] = m;
    }
    float mx = msg[0];
#pragma unroll
    for (int cj = 1; cj < 8; ++cj) mx = fmaxf(mx, msg[cj]);
    float e[8], sum = 0.f;
#pragma unroll
    for (int cj = 0; cj < 8; ++cj) { e[cj] = __expf(msg[cj] - mx); sum += e[cj]; }
    float inv = 0.5f / sum;  // DAMP * softmax
    const float* mo = mcur + (size_t)tid * 8;
    float* mn = mnext + (size_t)tid * 8;
#pragma unroll
    for (int cj = 0; cj < 8; ++cj)
        mn[cj] = __logf(e[cj] * inv + 0.5f * __expf(mo[cj]));
}

// ---------- final output ----------
__global__ __launch_bounds__(256) void k_out(const float* __restrict__ S,
                                             const float* __restrict__ mbar,
                                             float* __restrict__ out) {
    int i = blockIdx.x * blockDim.x + threadIdx.x;  // 0..1023
    float u[8], mx = -1e30f;
#pragma unroll
    for (int c = 0; c < 8; ++c) {
        u[c] = S[i * 8 + c] - mbar[((size_t)i * NN + i) * 8 + c];
        mx = fmaxf(mx, u[c]);
    }
    float e[8], sum = 0.f;
#pragma unroll
    for (int c = 0; c < 8; ++c) { e[c] = __expf(u[c] - mx); sum += e[c]; }
    float inv = 1.f / sum;
#pragma unroll
    for (int c = 0; c < 8; ++c) out[i * 8 + c] = e[c] * inv;
}

extern "C" void kernel_launch(void* const* d_in, const int* in_sizes, int n_in,
                              void* d_out, int out_size, void* d_ws, size_t ws_size,
                              hipStream_t stream) {
    const float* E  = (const float*)d_in[0];
    const float* F  = (const float*)d_in[1];
    const float* Fc = (const float*)d_in[2];
    const float* B  = (const float*)d_in[3];
    const float* R  = (const float*)d_in[4];
    const float* D  = (const float*)d_in[5];
    float* out = (float*)d_out;

    char* ws = (char*)d_ws;
    size_t off = 0;
    auto alloc = [&](size_t bytes) -> char* {
        char* p = ws + off;
        off += (bytes + 255) & ~(size_t)255;
        return p;
    };
    _Float16* Eh   = (_Float16*)alloc((size_t)MROW * DIMD * 2);          //   2 MB
    _Float16* Ah   = (_Float16*)alloc((size_t)3 * MROW * DIMD * 2);      //   6 MB
    _Float16* Fh   = (_Float16*)alloc((size_t)NN * DIMD * 2);            // 256 KB
    _Float16* FDh  = (_Float16*)alloc((size_t)3 * NN * DIMD * 2);        // 768 KB
    float*    psi  = (float*)alloc((size_t)MROW * 4);                    //  32 KB
    float*    aW   = (float*)alloc((size_t)NN * NN * 4 * 4);             //  16 MB
    float*    Sbuf = (float*)alloc((size_t)MROW * 4);                    //  32 KB
    _Float16* phi  = (_Float16*)alloc((size_t)MROW * MROW * 2);          // 128 MB
    float*    mbarA = (float*)alloc((size_t)NN * NN * 8 * 4);            //  32 MB
    float*    mbarB = (float*)alloc((size_t)NN * NN * 8 * 4);            //  32 MB

    // init + prep
    k_zero <<<8192, 256, 0, stream>>>(mbarA);                 // 8M floats
    k_prepE<<<4096, 256, 0, stream>>>(E, R, Eh, Ah);          // 1M elems
    k_prepF<<<512,  256, 0, stream>>>(F, D, Fh, FDh);         // 128K elems
    k_psi  <<<32,   256, 0, stream>>>(E, B, Fc, psi);         // 8192 outputs

    // a[i][j][k] via WMMA + softmax_k
    k_logits<<<128, 256, 0, stream>>>(FDh, Fh, aW);           // 1024 waves

    // phi via 3x WMMA GEMM + weighted combine
    k_phi<<<8192, 256, 0, stream>>>(Ah, Eh, aW, phi);         // 65536 waves

    // LBP loop (ping-pong mbar)
    float* cur = mbarA;
    float* nxt = mbarB;
    for (int it = 0; it < 10; ++it) {
        k_reduceS<<<256,  256, 0, stream>>>(cur, psi, Sbuf);
        k_update <<<4096, 256, 0, stream>>>(cur, Sbuf, phi, nxt);
        float* t = cur; cur = nxt; nxt = t;
    }

    // final marginals
    k_reduceS<<<256, 256, 0, stream>>>(cur, psi, Sbuf);
    k_out   <<<4,   256, 0, stream>>>(Sbuf, cur, out);
}
